// Attention_11802570130232
// MI455X (gfx1250) — matrix operand order repaired and checked by reading, not compiled, not run on hardware
//
#include <hip/hip_runtime.h>

#define BATCH 16
#define SEQL  1024
#define DMODEL 768

typedef __attribute__((ext_vector_type(8)))  float  v8f;
typedef __attribute__((ext_vector_type(16))) __bf16 v16bf;

// ---- gfx1250 async global->LDS path (ASYNCcnt-tracked), via inline asm -----
// (ROCm 7.2 declares __builtin_amdgcn_global_load_async_to_lds_b128 with an
//  undocumented prototype; inline asm bypasses the builtin signature.)
#define USE_ASYNC_LDS 1

static __device__ __forceinline__ void cp16_to_lds(unsigned short* l,
                                                   const unsigned short* g) {
#if USE_ASYNC_LDS
    unsigned int loff = (unsigned int)(unsigned long long)
        (__attribute__((address_space(3))) unsigned short*)l;   // LDS byte offset
    unsigned long long gaddr = (unsigned long long)g;           // global VA
    asm volatile("global_load_async_to_lds_b128 %0, %1, off"
                 :
                 : "v"(loff), "v"(gaddr)
                 : "memory");
#else
    *(uint4*)l = *(const uint4*)g;
#endif
}
static __device__ __forceinline__ void wait_async() {
#if USE_ASYNC_LDS
    asm volatile("s_wait_asynccnt 0x0" ::: "memory");
#endif
}

struct __align__(16) BfPack { uint4 lo, hi; };

// float -> bf16 round-to-nearest-even
static __device__ __forceinline__ unsigned short f2bf(float f) {
    unsigned int u = __float_as_uint(f);
    u += 0x7FFFu + ((u >> 16) & 1u);
    return (unsigned short)(u >> 16);
}

// A fragment, 16x32 bf16 (M x K). p points at (row = lane&15)'s K-window base.
// lanes 0-15: v0..3 = K 0..7, v4..7 = K 16..23 ; lanes 16-31: K 8..15 / 24..31.
static __device__ __forceinline__ v16bf frag_a(const unsigned short* p, int lane) {
    const int b1 = (lane & 16) ? 8 : 0;
    BfPack k;
    k.lo = *(const uint4*)(p + b1);
    k.hi = *(const uint4*)(p + b1 + 16);
    return __builtin_bit_cast(v16bf, k);
}

// B fragment, 32x16 bf16 (K x N). p points at (col = lane&15)'s K-window base
// (cols stored as contiguous-K rows). lanes 0-15: K 0..7 / 16..23, lanes 16-31: K 8..15 / 24..31.
static __device__ __forceinline__ v16bf frag_b(const unsigned short* p, int lane) {
    const int o = (lane & 16) ? 8 : 0;
    BfPack k;
    k.lo = *(const uint4*)(p + o);
    k.hi = *(const uint4*)(p + o + 16);
    return __builtin_bit_cast(v16bf, k);
}

// ---------------------------------------------------------------------------
// Kernel 0: add positional encoding, convert to bf16, emit x, q, and x^T.
// grid (DMODEL/64, SEQL/64, BATCH), 256 threads.
// ---------------------------------------------------------------------------
__global__ __launch_bounds__(256) void prep_kernel(
    const float* __restrict__ in, const float* __restrict__ qu,
    unsigned short* __restrict__ xb, unsigned short* __restrict__ qb,
    unsigned short* __restrict__ xt)
{
    __shared__ unsigned short tile[64 * 66];   // 66 = conflict-free transpose pad
    const int b  = blockIdx.z;
    const int l0 = blockIdx.y * 64;
    const int d0 = blockIdx.x * 64;
    const int tid = threadIdx.x;
    const float C = -0.0119926306926773f;      // -ln(10000)/768

#pragma unroll
    for (int it = 0; it < 16; ++it) {
        int idx = it * 256 + tid;
        int li = idx >> 6, di = idx & 63;
        int l = l0 + li, d = d0 + di;
        float ang = (float)l * __expf((float)(d & ~1) * C);
        float pe  = (d & 1) ? __cosf(ang) : __sinf(ang);
        size_t g = ((size_t)b * SEQL + l) * DMODEL + d;
        float xv = in[g] + pe;
        float qv = qu[g] + pe;
        xb[g] = f2bf(xv);
        qb[g] = f2bf(qv);
        tile[li * 66 + di] = f2bf(xv);
    }
    __syncthreads();
#pragma unroll
    for (int it = 0; it < 16; ++it) {
        int idx = it * 256 + tid;
        int dt = idx >> 6, lt = idx & 63;     // lt contiguous -> coalesced store
        xt[((size_t)b * DMODEL + (d0 + dt)) * SEQL + (l0 + lt)] = tile[lt * 66 + dt];
    }
}

// ---------------------------------------------------------------------------
// Kernel 1: scores = x q^T / sqrt(d), softmax over keys, write p_attn (f32).
// grid (SEQL/32, BATCH), 256 threads (8 waves). Each wave owns a 2x2 grid of
// 16x16 tiles (both row-tiles x 2 col-tiles) -> A/B fragment reuse cuts LDS
// reads to 2 x ds_load_b128 per WMMA. Key block = 256 cols, K-chunked staging.
// LDS: 48.5KB x-rows + 100KB key chunk + 1KB reduce (CDNA5: 320KB/WGP).
// ---------------------------------------------------------------------------
#define PADR 776   // 768 + 8 bf16 pad (row stride 1552B, 16B aligned)
#define PADQ 200   // 192 + 8 bf16 pad (row stride  400B, 16B aligned)

__global__ __launch_bounds__(256) void scores_softmax_kernel(
    const unsigned short* __restrict__ xb, const unsigned short* __restrict__ qb,
    float* __restrict__ p_out)
{
    __shared__ unsigned short sA[32 * PADR];      // 49,664 B
    __shared__ unsigned short sB[256 * PADQ];     // 102,400 B
    __shared__ float sRed[32 * 8];                // 1,024 B

    const int b  = blockIdx.y;
    const int l0 = blockIdx.x * 32;
    const int tid = threadIdx.x, w = tid >> 5, lane = tid & 31;
    const int rsel = lane >> 4, ln = lane & 15;
    const float scale = 0.03608439182435161f;     // 1/sqrt(768)

    // stage 32 query rows of x (bf16, K 0..191 of each row) via async DMA
    const unsigned short* xrow = xb + ((size_t)b * SEQL + l0) * DMODEL;
#pragma unroll
    for (int it = 0; it < 3; ++it) {
        int lin = it * 256 + tid;
        int r = lin / 24, c = lin % 24;           // 24 x 16B per row
        cp16_to_lds(sA + r * PADR + c * 8, xrow + r * DMODEL + c * 8);
    }

    v8f vz = {0.f, 0.f, 0.f, 0.f, 0.f, 0.f, 0.f, 0.f};
    v8f acc[16];                                  // [cb][rtile][ctile]
#pragma unroll
    for (int t = 0; t < 16; ++t) acc[t] = vz;

#pragma unroll
    for (int cb = 0; cb < 4; ++cb) {              // 4 key blocks of 256 cols
        const unsigned short* qbase = qb + ((size_t)b * SEQL + cb * 256) * DMODEL;
#pragma unroll
        for (int kh = 0; kh < 4; ++kh) {          // 4 K-chunks of 192
            __syncthreads();                      // prev readers done with sB
#pragma unroll
            for (int it = 0; it < 24; ++it) {
                int lin = it * 256 + tid;
                int r = lin / 24, c = lin % 24;
                cp16_to_lds(sB + r * PADQ + c * 8,
                            qbase + (size_t)r * DMODEL + kh * 192 + c * 8);
            }
            wait_async();
            __syncthreads();
            const unsigned short* a0p = sA + ln * PADR + kh * 192;
            const unsigned short* a1p = a0p + 16 * PADR;
            const unsigned short* b0p = sB + (w * 32 + ln) * PADQ;
            const unsigned short* b1p = b0p + 16 * PADQ;
            for (int k = 0; k < 6; ++k) {
                v16bf A0 = frag_a(a0p + k * 32, lane);
                v16bf A1 = frag_a(a1p + k * 32, lane);
                v16bf B0 = frag_b(b0p + k * 32, lane);
                v16bf B1 = frag_b(b1p + k * 32, lane);
                acc[cb * 4 + 0] = __builtin_amdgcn_wmma_f32_16x16x32_bf16(
                    false, A0, false, B0, (short)0, acc[cb * 4 + 0], false, false);
                acc[cb * 4 + 1] = __builtin_amdgcn_wmma_f32_16x16x32_bf16(
                    false, A0, false, B1, (short)0, acc[cb * 4 + 1], false, false);
                acc[cb * 4 + 2] = __builtin_amdgcn_wmma_f32_16x16x32_bf16(
                    false, A1, false, B0, (short)0, acc[cb * 4 + 2], false, false);
                acc[cb * 4 + 3] = __builtin_amdgcn_wmma_f32_16x16x32_bf16(
                    false, A1, false, B1, (short)0, acc[cb * 4 + 3], false, false);
            }
        }
    }

    // ---- softmax (row slot s = rtile*8 + j ; global row = rtile*16+j+8*rsel) ---
    float rmax[16];
#pragma unroll
    for (int s = 0; s < 16; ++s) rmax[s] = -3.402823466e38f;
#pragma unroll
    for (int cb = 0; cb < 4; ++cb)
#pragma unroll
        for (int rtile = 0; rtile < 2; ++rtile)
#pragma unroll
            for (int ct = 0; ct < 2; ++ct)
#pragma unroll
                for (int j = 0; j < 8; ++j)
                    rmax[rtile * 8 + j] =
                        fmaxf(rmax[rtile * 8 + j], acc[cb * 4 + rtile * 2 + ct][j]);
#pragma unroll
    for (int m = 1; m < 16; m <<= 1)
#pragma unroll
        for (int s = 0; s < 16; ++s)
            rmax[s] = fmaxf(rmax[s], __shfl_xor(rmax[s], m, 32));

    if (ln == 0) {
#pragma unroll
        for (int rtile = 0; rtile < 2; ++rtile)
#pragma unroll
            for (int j = 0; j < 8; ++j)
                sRed[(rtile * 16 + rsel * 8 + j) * 8 + w] = rmax[rtile * 8 + j];
    }
    __syncthreads();
#pragma unroll
    for (int rtile = 0; rtile < 2; ++rtile)
#pragma unroll
        for (int j = 0; j < 8; ++j) {
            float m = -3.402823466e38f;
#pragma unroll
            for (int q = 0; q < 8; ++q)
                m = fmaxf(m, sRed[(rtile * 16 + rsel * 8 + j) * 8 + q]);
            rmax[rtile * 8 + j] = m;
        }
    __syncthreads();

    float rsum[16];
#pragma unroll
    for (int s = 0; s < 16; ++s) rsum[s] = 0.f;
#pragma unroll
    for (int cb = 0; cb < 4; ++cb)
#pragma unroll
        for (int rtile = 0; rtile < 2; ++rtile)
#pragma unroll
            for (int ct = 0; ct < 2; ++ct)
#pragma unroll
                for (int j = 0; j < 8; ++j) {
                    int s = rtile * 8 + j;
                    float e = __expf((acc[cb * 4 + rtile * 2 + ct][j] - rmax[s]) * scale);
                    acc[cb * 4 + rtile * 2 + ct][j] = e;
                    rsum[s] += e;
                }
#pragma unroll
    for (int m = 1; m < 16; m <<= 1)
#pragma unroll
        for (int s = 0; s < 16; ++s) rsum[s] += __shfl_xor(rsum[s], m, 32);
    if (ln == 0) {
#pragma unroll
        for (int rtile = 0; rtile < 2; ++rtile)
#pragma unroll
            for (int j = 0; j < 8; ++j)
                sRed[(rtile * 16 + rsel * 8 + j) * 8 + w] = rsum[rtile * 8 + j];
    }
    __syncthreads();
#pragma unroll
    for (int rtile = 0; rtile < 2; ++rtile)
#pragma unroll
        for (int j = 0; j < 8; ++j) {
            float s = 0.f;
#pragma unroll
            for (int q = 0; q < 8; ++q)
                s += sRed[(rtile * 16 + rsel * 8 + j) * 8 + q];
            rsum[rtile * 8 + j] = 1.0f / s;
        }

    float* prow = p_out + ((size_t)b * SEQL + l0) * SEQL;
#pragma unroll
    for (int cb = 0; cb < 4; ++cb)
#pragma unroll
        for (int rtile = 0; rtile < 2; ++rtile)
#pragma unroll
            for (int ct = 0; ct < 2; ++ct)
#pragma unroll
                for (int j = 0; j < 8; ++j) {
                    int row = rtile * 16 + j + rsel * 8;
                    int col = cb * 256 + w * 32 + ct * 16 + ln;
                    prow[(size_t)row * SEQL + col] =
                        acc[cb * 4 + rtile * 2 + ct][j] * rsum[rtile * 8 + j];
                }
}

// ---------------------------------------------------------------------------
// Kernel 2: out = p_attn @ x.  A = p (f32 -> bf16 on stage), B = x^T rows
// (async-DMA'd into LDS). A-frag reused across 12 col-tiles per wave.
// grid (SEQL/32, BATCH), 256 threads. LDS: 2.5KB + 60KB.
// ---------------------------------------------------------------------------
#define PADK 40    // 32 + 8 bf16 pad (80B row stride, 16B aligned)

__global__ __launch_bounds__(256) void attn_pv_kernel(
    const float* __restrict__ p_in, const unsigned short* __restrict__ xt,
    float* __restrict__ out)
{
    __shared__ unsigned short sA2[32 * PADK];
    __shared__ unsigned short sB2[DMODEL * PADK];

    const int b  = blockIdx.y;
    const int l0 = blockIdx.x * 32;
    const int tid = threadIdx.x, w = tid >> 5, lane = tid & 31;
    const int rt = w >> 2, cg = w & 3;
    const int rsel = lane >> 4, ln = lane & 15;

    v8f vz = {0.f, 0.f, 0.f, 0.f, 0.f, 0.f, 0.f, 0.f};
    v8f acc[12];
#pragma unroll
    for (int t = 0; t < 12; ++t) acc[t] = vz;

    const float* pbase = p_in + ((size_t)b * SEQL + l0) * SEQL;
    const unsigned short* xtb = xt + (size_t)b * DMODEL * SEQL;

    for (int kc = 0; kc < 32; ++kc) {
        const int k0 = kc * 32;
        __syncthreads();
        // stage A: 32x32 block of p, f32 -> bf16 (converted in registers)
#pragma unroll
        for (int it = 0; it < 4; ++it) {
            int lin = it * 256 + tid;
            int r = lin >> 5, c = lin & 31;
            sA2[r * PADK + c] = f2bf(pbase[(size_t)r * SEQL + k0 + c]);
        }
        // stage B: x^T rows d=0..767, 32 K-values each -> async DMA to LDS
#pragma unroll
        for (int it = 0; it < 12; ++it) {
            int lin = it * 256 + tid;
            int r = lin >> 2, c = lin & 3;
            cp16_to_lds(sB2 + r * PADK + c * 8,
                        xtb + (size_t)r * SEQL + k0 + c * 8);
        }
        wait_async();
        __syncthreads();
        v16bf a = frag_a(sA2 + (rt * 16 + ln) * PADK, lane);
#pragma unroll
        for (int ct = 0; ct < 12; ++ct) {
            v16bf bf = frag_b(sB2 + (cg * 192 + ct * 16 + ln) * PADK, lane);
            acc[ct] = __builtin_amdgcn_wmma_f32_16x16x32_bf16(
                false, a, false, bf, (short)0, acc[ct], false, false);
        }
    }

    float* obase = out + ((size_t)b * SEQL + l0) * DMODEL;
#pragma unroll
    for (int ct = 0; ct < 12; ++ct)
#pragma unroll
        for (int j = 0; j < 8; ++j) {
            int row = rt * 16 + j + rsel * 8;
            int col = cg * 192 + ct * 16 + ln;
            obase[(size_t)row * DMODEL + col] = acc[ct][j];
        }
}

// ---------------------------------------------------------------------------
extern "C" void kernel_launch(void* const* d_in, const int* in_sizes, int n_in,
                              void* d_out, int out_size, void* d_ws, size_t ws_size,
                              hipStream_t stream)
{
    (void)in_sizes; (void)n_in; (void)out_size; (void)ws_size;
    const float* inputs    = (const float*)d_in[0];
    const float* questions = (const float*)d_in[1];

    float* out    = (float*)d_out;                                   // [16,1024,768]
    float* p_attn = out + (size_t)BATCH * SEQL * DMODEL;             // [16,1024,1024]

    // workspace: x_bf16 | q_bf16 | xT_bf16  (24 MB each)
    unsigned short* xb = (unsigned short*)d_ws;
    unsigned short* qb = xb + (size_t)BATCH * SEQL * DMODEL;
    unsigned short* xt = qb + (size_t)BATCH * SEQL * DMODEL;

    prep_kernel<<<dim3(DMODEL / 64, SEQL / 64, BATCH), 256, 0, stream>>>(
        inputs, questions, xb, qb, xt);
    scores_softmax_kernel<<<dim3(SEQL / 32, BATCH), 256, 0, stream>>>(
        xb, qb, p_attn);
    attn_pv_kernel<<<dim3(SEQL / 32, BATCH), 256, 0, stream>>>(
        p_attn, xt, out);
}
